// Coarse_layer_90374701842959
// MI455X (gfx1250) — matrix operand order repaired and checked by reading, not compiled, not run on hardware
//
#include <hip/hip_runtime.h>
#include <hip/hip_bf16.h>
#include <stdint.h>

// ---------------------------------------------------------------------------
// Coarse_layer on MI455X (gfx1250): memory-bound graph coarsening.
//   out = [ node_avg (Nc x 128 f32) | edge_avg (Ec x 128 f32) | edge_index raw ]
// GEMM path uses bf16x3 split-precision WMMA (v_wmma_f32_16x16x32_bf16),
// weights staged in LDS in B-fragment order. Scatters use f32 global atomics.
// ---------------------------------------------------------------------------

#define HD 128  // hidden dim (reference H)

typedef __attribute__((ext_vector_type(16))) __bf16 v16bf;
typedef __attribute__((ext_vector_type(8)))  float  v8f;

// ---------------- fragment layout helpers -----------------------------------
// Packed weight fragments: [layer(2)][j(8)][s(4)][var(2: hi,lo)][lane(32)][e(16)] bf16
// Per (layer,j,s): 2048 bytes. var stride 1024 B. lane stride 32 B.
__device__ __host__ inline int frag_byte_off(int layer, int j, int s, int var, int lane) {
    return (((layer * 8 + j) * 4 + s) * 2048) + var * 1024 + lane * 32;
}

// ---------------- 1) pack W1/W2 into bf16 hi/lo B-fragments -----------------
__global__ __launch_bounds__(256)
void pack_weights_kernel(const float* __restrict__ W1, const float* __restrict__ W2,
                         __bf16* __restrict__ wpack) {
    int i = blockIdx.x * blockDim.x + threadIdx.x;   // 32768 positions
    if (i >= 2 * 8 * 4 * 32 * 16) return;
    int e     =  i        & 15;
    int lane  = (i >> 4)  & 31;
    int s     = (i >> 9)  & 3;
    int j     = (i >> 11) & 7;
    int layer = (i >> 14) & 1;
    // B (32x16 bf16) contiguous k order: cols = lane&15; lanes>=16 hold K+16; elem e -> K base+e
    int k   = 32 * s + 16 * (lane >> 4) + e;
    int col = 16 * j + (lane & 15);
    float v = layer ? W2[k * HD + col] : W1[k * HD + col];
    __bf16 hi = (__bf16)v;
    __bf16 lo = (__bf16)(v - (float)hi);
    __bf16* base = (__bf16*)((char*)wpack + frag_byte_off(layer, j, s, 0, lane));
    base[e]       = hi;          // var 0
    base[512 + e] = lo;          // var 1 (+1024 B)
}

// ---------------- 2) edge scatter-add (1 wave per fine edge row) ------------
__global__ __launch_bounds__(256)
void edge_scatter_kernel(const float* __restrict__ ea, const void* __restrict__ f2ce,
                         int idx64, float* __restrict__ edge_out,
                         float* __restrict__ ecnt, int e_fine) {
    int gw   = (blockIdx.x * blockDim.x + threadIdx.x) >> 5;
    int lane = threadIdx.x & 31;
    if (gw >= e_fine) return;
    long long tgt = idx64 ? ((const long long*)f2ce)[gw] : (long long)((const int*)f2ce)[gw];
    const float4 v = ((const float4*)(ea + (size_t)gw * HD))[lane];
    float* dst = edge_out + (size_t)tgt * HD + lane * 4;
    atomicAdd(dst + 0, v.x);
    atomicAdd(dst + 1, v.y);
    atomicAdd(dst + 2, v.z);
    atomicAdd(dst + 3, v.w);
    if (lane == 0) atomicAdd(ecnt + tgt, 1.0f);
}

// ---------------- 3) per-fine-node count ------------------------------------
__global__ __launch_bounds__(256)
void node_count_kernel(const void* __restrict__ f2ci, int idx64,
                       float* __restrict__ ncnt, int n_fine) {
    int i = blockIdx.x * blockDim.x + threadIdx.x;
    if (i >= n_fine) return;
    long long c = idx64 ? ((const long long*)f2ci)[i] : (long long)((const int*)f2ci)[i];
    atomicAdd(ncnt + c, 1.0f);
}

// ---------------- 4) sums -> means (in place) -------------------------------
__global__ __launch_bounds__(256)
void div_kernel(float* __restrict__ buf, const float* __restrict__ cnt, int n_rows) {
    int i = blockIdx.x * blockDim.x + threadIdx.x;       // one float4 per thread
    if (i >= n_rows * (HD / 4)) return;
    int r = i >> 5;
    float c = cnt[r];
    float inv = 1.0f / (c > 1.0f ? c : 1.0f);
    float4* p = (float4*)buf + i;
    float4 v = *p;
    v.x *= inv; v.y *= inv; v.z *= inv; v.w *= inv;
    *p = v;
}

// ---------------- 5) MLP (bf16x3 WMMA) + node scatter -----------------------
// Block = 128 threads (4 waves). Dynamic LDS:
//   [0, 128KB)         packed weight fragments (copied from global once)
//   [128KB, 160KB)     per-wave h staging: 4 * 16*128 f32
__global__ __launch_bounds__(128)
void mlp_scatter_kernel(const float* __restrict__ x, const float* __restrict__ dist,
                        const void* __restrict__ f2ci, int idx64,
                        const float* __restrict__ W1, const float* __restrict__ b1,
                        const float* __restrict__ b2,
                        const float* __restrict__ wpack,
                        float* __restrict__ node_out, int n_fine) {
    extern __shared__ char smem[];
    float* hstage = (float*)(smem + 131072);

    const int tid  = threadIdx.x;
    const int wave = tid >> 5;
    const int lane = tid & 31;
    const int m    = lane & 15;
    const int half = lane >> 4;

    // cooperative copy of packed fragments: 128 KB = 32768 dwords
    {
        const uint32_t* src = (const uint32_t*)wpack;
        uint32_t* dst = (uint32_t*)smem;
        for (int i = tid; i < 32768; i += blockDim.x) dst[i] = src[i];
    }
    __syncthreads();

    float* hs = hstage + wave * (16 * HD);
    const int nTiles = n_fine >> 4;                       // rows per tile = 16
    const int wavesTotal = gridDim.x * (blockDim.x >> 5);
    for (int t = blockIdx.x * (blockDim.x >> 5) + wave; t < nTiles; t += wavesTotal) {
        const int rowbase = t << 4;

        // prefetch next tile's x row for this lane (global_prefetch_b8)
        {
            int tn = t + wavesTotal;
            if (tn < nTiles) {
                const char* p = (const char*)(x + (size_t)((tn << 4) + m) * HD);
                __builtin_prefetch(p, 0, 3);
                __builtin_prefetch(p + 256, 0, 3);
            }
        }

        // ---- A1 fragments from x (16x32, contiguous k = 16*half + elem, same order as B) ----
        v16bf a_hi[4], a_lo[4];
        const float* xrow = x + (size_t)(rowbase + m) * HD;
        #pragma unroll
        for (int s = 0; s < 4; ++s) {
            #pragma unroll
            for (int run = 0; run < 2; ++run) {
                const int kb = 32 * s + 16 * half + 8 * run;
                #pragma unroll
                for (int e = 0; e < 8; ++e) {
                    float v = xrow[kb + e];
                    __bf16 hi = (__bf16)v;
                    a_hi[s][run * 8 + e] = hi;
                    a_lo[s][run * 8 + e] = (__bf16)(v - (float)hi);
                }
            }
        }
        // distances for the 8 C-rows this lane owns (M = half? r+8 : r)
        float dv[8];
        #pragma unroll
        for (int r = 0; r < 8; ++r) dv[r] = dist[rowbase + (half ? r + 8 : r)];

        // ---- layer 1: h = relu([x|d] @ W1 + b1) ----
        for (int j = 0; j < 8; ++j) {
            const int col = 16 * j + m;
            const float b1v = b1[col];
            const float wlast = W1[128 * HD + col];       // row 128 of W1 (distance column)
            v8f c = {};
            #pragma unroll
            for (int r = 0; r < 8; ++r) c[r] = b1v + dv[r] * wlast;
            #pragma unroll
            for (int s = 0; s < 4; ++s) {
                v16bf Bh = *(const v16bf*)(smem + frag_byte_off(0, j, s, 0, lane));
                v16bf Bl = *(const v16bf*)(smem + frag_byte_off(0, j, s, 1, lane));
                c = __builtin_amdgcn_wmma_f32_16x16x32_bf16(false, a_hi[s], false, Bh, (short)0, c, false, false);
                c = __builtin_amdgcn_wmma_f32_16x16x32_bf16(false, a_hi[s], false, Bl, (short)0, c, false, false);
                c = __builtin_amdgcn_wmma_f32_16x16x32_bf16(false, a_lo[s], false, Bh, (short)0, c, false, false);
            }
            #pragma unroll
            for (int r = 0; r < 8; ++r) {
                float v = c[r] > 0.0f ? c[r] : 0.0f;      // relu
                hs[(half ? r + 8 : r) * HD + col] = v;    // stage for layer 2 (same-wave DS: in order)
            }
        }

        // ---- A2 fragments from staged h ----
        v16bf h_hi[4], h_lo[4];
        const float* hrow = hs + m * HD;
        #pragma unroll
        for (int s = 0; s < 4; ++s) {
            #pragma unroll
            for (int run = 0; run < 2; ++run) {
                const int kb = 32 * s + 16 * half + 8 * run;
                #pragma unroll
                for (int e = 0; e < 8; ++e) {
                    float v = hrow[kb + e];
                    __bf16 hi = (__bf16)v;
                    h_hi[s][run * 8 + e] = hi;
                    h_lo[s][run * 8 + e] = (__bf16)(v - (float)hi);
                }
            }
        }
        // coarse node ids -> output row byte bases, for this lane's 8 rows
        float* outrow[8];
        #pragma unroll
        for (int r = 0; r < 8; ++r) {
            int row = rowbase + (half ? r + 8 : r);
            long long c = idx64 ? ((const long long*)f2ci)[row] : (long long)((const int*)f2ci)[row];
            outrow[r] = node_out + (size_t)c * HD;
        }

        // ---- layer 2 + residual + scatter-add ----
        for (int j = 0; j < 8; ++j) {
            const int col = 16 * j + m;
            const float b2v = b2[col];
            v8f c = {};
            #pragma unroll
            for (int r = 0; r < 8; ++r) c[r] = b2v;
            #pragma unroll
            for (int s = 0; s < 4; ++s) {
                v16bf Bh = *(const v16bf*)(smem + frag_byte_off(1, j, s, 0, lane));
                v16bf Bl = *(const v16bf*)(smem + frag_byte_off(1, j, s, 1, lane));
                c = __builtin_amdgcn_wmma_f32_16x16x32_bf16(false, h_hi[s], false, Bh, (short)0, c, false, false);
                c = __builtin_amdgcn_wmma_f32_16x16x32_bf16(false, h_hi[s], false, Bl, (short)0, c, false, false);
                c = __builtin_amdgcn_wmma_f32_16x16x32_bf16(false, h_lo[s], false, Bh, (short)0, c, false, false);
            }
            // batched residual loads first (8 loads in flight, one wait), then atomics
            float xv[8];
            #pragma unroll
            for (int r = 0; r < 8; ++r) {
                const int row = rowbase + (half ? r + 8 : r);
                xv[r] = x[(size_t)row * HD + col];
            }
            #pragma unroll
            for (int r = 0; r < 8; ++r) {
                atomicAdd(outrow[r] + col, xv[r] + c[r]);   // residual + segment sum
            }
        }
    }
}

// ---------------------------------------------------------------------------
extern "C" void kernel_launch(void* const* d_in, const int* in_sizes, int n_in,
                              void* d_out, int out_size, void* d_ws, size_t ws_size,
                              hipStream_t stream) {
    const float* x    = (const float*)d_in[0];
    const void*  cei  = d_in[1];                 // coarse_edge_index (pass-through)
    const float* ea   = (const float*)d_in[2];
    const void*  f2ci = d_in[3];
    const void*  f2ce = d_in[4];
    const float* dist = (const float*)d_in[5];
    const float* W1   = (const float*)d_in[6];
    const float* b1   = (const float*)d_in[7];
    const float* W2   = (const float*)d_in[8];
    const float* b2   = (const float*)d_in[9];

    const int n_fine   = in_sizes[0] / HD;       // 400000
    const int e_coarse = in_sizes[1] / 2;        // 400000
    const int e_fine   = in_sizes[2] / HD;       // 1600000
    const int n_coarse = 100000;                 // reference module constant

    float* out      = (float*)d_out;
    float* node_out = out;                                   // Nc * 128
    float* edge_out = out + (size_t)n_coarse * HD;           // Ec * 128
    const long long fbase = (long long)(n_coarse + e_coarse) * HD;
    const long long R = (long long)out_size - fbase;         // edge_index tail (f32 slots)
    const int idx64 = (R == 4LL * e_coarse) ? 1 : 0;         // int64 if 8B/elem, else int32

    // workspace: [ecnt Ec f32][ncnt Nc f32][wpack 128KB bf16 frags]
    float*  ecnt  = (float*)d_ws;
    float*  ncnt  = ecnt + e_coarse;
    __bf16* wpack = (__bf16*)(ncnt + n_coarse);

    // zero accumulators + counts (capture-safe async memset)
    hipMemsetAsync(d_out, 0, (size_t)fbase * 4, stream);
    hipMemsetAsync(d_ws, 0, ((size_t)e_coarse + n_coarse) * 4, stream);
    // pass-through edge index, raw bits
    if (R > 0)
        hipMemcpyAsync(out + fbase, cei, (size_t)R * 4, hipMemcpyDeviceToDevice, stream);

    // 1) pack weight fragments (32768 positions)
    pack_weights_kernel<<<128, 256, 0, stream>>>(W1, W2, wpack);

    // 2) edge scatter-add: 1 wave per fine edge
    {
        int waves_per_blk = 8;
        int blocks = (e_fine + waves_per_blk - 1) / waves_per_blk;
        edge_scatter_kernel<<<blocks, 256, 0, stream>>>(ea, f2ce, idx64, edge_out, ecnt, e_fine);
    }
    // 3) node counts
    node_count_kernel<<<(n_fine + 255) / 256, 256, 0, stream>>>(f2ci, idx64, ncnt, n_fine);

    // 4) MLP + node scatter (4 waves/block, 1 tile of 16 rows per wave)
    {
        const int tiles = n_fine >> 4;
        const int blocks = (tiles + 3) / 4;
        const size_t lds = 131072 + 4 * 16 * HD * sizeof(float);   // 160 KB
        hipFuncSetAttribute((const void*)mlp_scatter_kernel,
                            hipFuncAttributeMaxDynamicSharedMemorySize, (int)lds);
        mlp_scatter_kernel<<<blocks, 128, lds, stream>>>(x, dist, f2ci, idx64,
                                                         W1, b1, b2, (const float*)wpack,
                                                         node_out, n_fine);
    }

    // 5) sums -> means
    div_kernel<<<((size_t)e_coarse * 32 + 255) / 256, 256, 0, stream>>>(edge_out, ecnt, e_coarse);
    div_kernel<<<((size_t)n_coarse * 32 + 255) / 256, 256, 0, stream>>>(node_out, ncnt, n_coarse);
}
